// AttentionBlock_87522843559762
// MI455X (gfx1250) — matrix operand order repaired and checked by reading, not compiled, not run on hardware
//
#include <hip/hip_runtime.h>

// ---------------------------------------------------------------------------
// AttentionBlock on MI455X (gfx1250, wave32, WMMA f32_16x16x32_f16)
// B=2, H=W=64 (N=4096), C=512, GROUPS=32 (16 ch/group), EPS=1e-5
// ~86 GFLOP vs ~40MB HBM -> compute-bound at HBM level; attention made
// L2-friendly via per-workgroup LDS staging of K/V tiles (8 waves share each
// 64-key tile -> 8x L2 traffic cut, using CDNA5's 320KB/WGP LDS). Staging
// uses GLOBAL_LOAD_ASYNC_TO_LDS (ASYNCcnt) when the toolchain exposes it.
// All matmuls via v_wmma_f32_16x16x32_f16 (f32 accum).
// ---------------------------------------------------------------------------

typedef _Float16 f16;
typedef __attribute__((ext_vector_type(16))) _Float16 v16h;
typedef __attribute__((ext_vector_type(8)))  _Float16 v8h;
typedef __attribute__((ext_vector_type(4)))  _Float16 v4h;
typedef __attribute__((ext_vector_type(8)))  float    v8f;
typedef __attribute__((ext_vector_type(4)))  int      v4i;

#define NB 2
#define NN 4096
#define NC 512
#define NG 32
#define EPSV 1e-5f
#define QSCALE 0.04419417382415922f  // 1/sqrt(512)

#if __has_builtin(__builtin_amdgcn_global_load_async_to_lds_b128) && \
    __has_builtin(__builtin_amdgcn_s_wait_asynccnt)
#define USE_ASYNC_LDS 1
#define AS1 __attribute__((address_space(1)))
#define AS3 __attribute__((address_space(3)))
#else
#define USE_ASYNC_LDS 0
#endif

// --- WMMA fragment helpers --------------------------------------------------
// A fragment 16x32 f16 as loaded here (contiguous k, not the ISA table order): lane l holds row m=l%16; with
// h=l/16 the 16 halves are K = 16h+0..15 (two 16B chunks, matches B).
static __device__ inline v16h load_afrag(const f16* rowk0, int h) {
  v8h lo = *(const v8h*)(rowk0 + 16 * h);
  v8h hi = *(const v8h*)(rowk0 + 16 * h + 8);
  return __builtin_shufflevector(lo, hi, 0, 1, 2, 3, 4, 5, 6, 7,
                                 8, 9, 10, 11, 12, 13, 14, 15);
}
// B fragment 32x16 f16 as loaded here (same contiguous k as A): lane l holds col n=l%16, K = 16*h + 0..15
// (one contiguous 32B chunk) -> B source must be [n][k] row-major (i.e. W^T).
static __device__ inline v16h load_bfrag(const f16* colk0, int h) {
  return *(const v16h*)(colk0 + 16 * h);
}
static __device__ inline v8f wmma_f16(v16h a, v16h b, v8f c) {
  return __builtin_amdgcn_wmma_f32_16x16x32_f16(false, a, false, b,
                                                (short)0, c, false, false);
}
// C/D layout: VGPR j, lane l -> D[m = j + 8*(l/16)][n = l%16]

// --- half-wave (16-lane) reductions via ds_swizzle (rows live in 16 lanes) --
static __device__ inline float red16_max(float v) {
  v = fmaxf(v, __int_as_float(__builtin_amdgcn_ds_swizzle(__float_as_int(v), 0x041f)));
  v = fmaxf(v, __int_as_float(__builtin_amdgcn_ds_swizzle(__float_as_int(v), 0x081f)));
  v = fmaxf(v, __int_as_float(__builtin_amdgcn_ds_swizzle(__float_as_int(v), 0x101f)));
  v = fmaxf(v, __int_as_float(__builtin_amdgcn_ds_swizzle(__float_as_int(v), 0x201f)));
  return v;
}
static __device__ inline float red16_sum(float v) {
  v += __int_as_float(__builtin_amdgcn_ds_swizzle(__float_as_int(v), 0x041f));
  v += __int_as_float(__builtin_amdgcn_ds_swizzle(__float_as_int(v), 0x081f));
  v += __int_as_float(__builtin_amdgcn_ds_swizzle(__float_as_int(v), 0x101f));
  v += __int_as_float(__builtin_amdgcn_ds_swizzle(__float_as_int(v), 0x201f));
  return v;
}

// ---------------------------------------------------------------------------
// K0: weights -> f16, transposed: wT[c_out][c_in] = w[c_in][c_out]
// ---------------------------------------------------------------------------
__global__ __launch_bounds__(256) void prep_weights_kernel(
    const float* __restrict__ wq, const float* __restrict__ wk,
    const float* __restrict__ wv, const float* __restrict__ wp,
    f16* __restrict__ wT) {
  unsigned idx = blockIdx.x * 256u + threadIdx.x;  // 4 * 512 * 512 total
  unsigned m = idx >> 18;
  unsigned rem = idx & 262143u;
  unsigned i = rem >> 9;    // c_in
  unsigned o = rem & 511u;  // c_out
  const float* src = (m == 0) ? wq : ((m == 1) ? wk : ((m == 2) ? wv : wp));
  wT[(size_t)m * 262144u + (size_t)o * 512u + i] = (f16)src[rem];
}

// ---------------------------------------------------------------------------
// K1: GroupNorm stats: one block per (b, g); reduce 4096*16 elements
// ---------------------------------------------------------------------------
__global__ __launch_bounds__(256) void gn_stats_kernel(
    const float* __restrict__ x, float* __restrict__ stats) {
  const int bg = blockIdx.x;
  const int b = bg >> 5, g = bg & 31;
  const float* base = x + (size_t)b * NN * NC + g * 16;
  float s = 0.f, ss = 0.f;
  for (int n = threadIdx.x; n < NN; n += 256) {
    const float4* p = (const float4*)(base + (size_t)n * NC);
#pragma unroll
    for (int i = 0; i < 4; ++i) {
      float4 v = p[i];
      s += v.x + v.y + v.z + v.w;
      ss += v.x * v.x + v.y * v.y + v.z * v.z + v.w * v.w;
    }
  }
  __shared__ float sh[256], sh2[256];
  sh[threadIdx.x] = s;
  sh2[threadIdx.x] = ss;
  __syncthreads();
  for (int o = 128; o > 0; o >>= 1) {
    if ((int)threadIdx.x < o) {
      sh[threadIdx.x] += sh[threadIdx.x + o];
      sh2[threadIdx.x] += sh2[threadIdx.x + o];
    }
    __syncthreads();
  }
  if (threadIdx.x == 0) {
    const float inv = 1.f / 65536.f;
    float mean = sh[0] * inv;
    float var = sh2[0] * inv - mean * mean;
    stats[bg * 2 + 0] = mean;
    stats[bg * 2 + 1] = rsqrtf(var + EPSV);
  }
}

// ---------------------------------------------------------------------------
// K2: apply GN, emit hn as f16 [b][n][c]
// ---------------------------------------------------------------------------
__global__ __launch_bounds__(256) void gn_apply_kernel(
    const float* __restrict__ x, const float* __restrict__ gamma,
    const float* __restrict__ beta, const float* __restrict__ stats,
    f16* __restrict__ hn) {
  size_t e = ((size_t)blockIdx.x * 256 + threadIdx.x) * 4;
  int c = (int)(e & (NC - 1));
  int b = (int)(e >> 21);  // N*C = 2^21
  int g = c >> 4;
  float mean = stats[(b * NG + g) * 2 + 0];
  float rstd = stats[(b * NG + g) * 2 + 1];
  float4 xv = *(const float4*)(x + e);
  float4 gv = *(const float4*)(gamma + c);
  float4 bv = *(const float4*)(beta + c);
  v4h o;
  o[0] = (f16)(((xv.x - mean) * rstd) * gv.x + bv.x);
  o[1] = (f16)(((xv.y - mean) * rstd) * gv.y + bv.y);
  o[2] = (f16)(((xv.z - mean) * rstd) * gv.z + bv.z);
  o[3] = (f16)(((xv.w - mean) * rstd) * gv.w + bv.w);
  *(v4h*)(hn + e) = o;
}

// ---------------------------------------------------------------------------
// K3: QKV GEMM. One wave per 16x16 output tile, 16 WMMAs over K=512.
//   q,k stored [b][n][c] f16 (q pre-scaled by 1/sqrt(C)); v stored transposed
//   vt[b][c][n] f16 so the P*V B-fragment load is contiguous.
// ---------------------------------------------------------------------------
__global__ __launch_bounds__(256) void qkv_gemm_kernel(
    const f16* __restrict__ hn, const f16* __restrict__ wTq,
    const f16* __restrict__ wTk, const f16* __restrict__ wTv,
    const float* __restrict__ bq, const float* __restrict__ bk,
    const float* __restrict__ bv, f16* __restrict__ q, f16* __restrict__ k,
    f16* __restrict__ vt) {
  const int tid = threadIdx.x, lane = tid & 31, r = lane & 15, h = lane >> 4;
  unsigned wid = blockIdx.x * 8u + (tid >> 5);
  const int ct = wid & 31; wid >>= 5;    // channel tile (n)
  const int mt = wid & 255; wid >>= 8;   // token tile (m)
  const int mat = wid % 3;
  const int b = wid / 3;
  const f16* wT = (mat == 0) ? wTq : ((mat == 1) ? wTk : wTv);
  const float* bias = (mat == 0) ? bq : ((mat == 1) ? bk : bv);
  const f16* arow = hn + (size_t)(b * NN + mt * 16 + r) * NC;
  const f16* brow = wT + (size_t)(ct * 16 + r) * NC;
  v8f acc = {0.f, 0.f, 0.f, 0.f, 0.f, 0.f, 0.f, 0.f};
#pragma unroll
  for (int kk = 0; kk < 16; ++kk) {
    v16h a = load_afrag(arow + kk * 32, h);
    v16h bf = load_bfrag(brow + kk * 32, h);
    acc = wmma_f16(a, bf, acc);
  }
  const float bn = bias[ct * 16 + r];
  if (mat == 0) {
#pragma unroll
    for (int j = 0; j < 8; ++j)
      q[(size_t)(b * NN + mt * 16 + j + 8 * h) * NC + ct * 16 + r] =
          (f16)((acc[j] + bn) * QSCALE);
  } else if (mat == 1) {
#pragma unroll
    for (int j = 0; j < 8; ++j)
      k[(size_t)(b * NN + mt * 16 + j + 8 * h) * NC + ct * 16 + r] =
          (f16)(acc[j] + bn);
  } else {
#pragma unroll
    for (int j = 0; j < 8; ++j)
      vt[(size_t)(b * NC + ct * 16 + r) * NN + mt * 16 + j + 8 * h] =
          (f16)(acc[j] + bn);
  }
}

// ---------------------------------------------------------------------------
// K4: Flash attention with workgroup-cooperative LDS staging.
//   Block = 8 waves (4 query tiles x 2 channel halves, same batch). Per
//   64-key tile: all 256 threads stage K (64x512) and V^T (512x64) into LDS
//   once (vs 8x redundant L2 reads) -- via async global->LDS DMA (ASYNCcnt)
//   when available -- then each wave runs 64 WMMAs for S = Q*K^T, online
//   softmax (ds_swizzle reductions), P via LDS to A-fragments, 32 WMMAs for
//   P*V. Row pitches padded +16B vs the 256B bank period to avoid conflicts.
// ---------------------------------------------------------------------------
#define BC 64
#define KPAD 520  // 512 + 8 halves (1040B pitch)
#define VPAD 72   // 64 + 8 halves  (144B pitch)
#define PPAD 72

__global__ __launch_bounds__(256) void attn_kernel(
    const f16* __restrict__ qb, const f16* __restrict__ kb,
    const f16* __restrict__ vt, f16* __restrict__ ob) {
  __shared__ f16 Klds[BC * KPAD];     // 66560 B, [key][c] padded
  __shared__ f16 Vlds[NC * VPAD];     // 73728 B, [c][key] padded
  __shared__ f16 Pbuf[8][16 * PPAD];  // 18432 B, per-wave P tile
  const int tid = threadIdx.x, lane = tid & 31, r = lane & 15, h = lane >> 4;
  const int wslot = tid >> 5;
  const unsigned wid = blockIdx.x * 8u + wslot;
  const int half = wid & 1;         // channel half: 0 -> ch 0..255
  const int qt = (wid >> 1) & 255;  // query tile
  const int b = wid >> 9;
  const int m0 = qt * 16;
  const int ch0 = half * 256;

  const f16* qrow = qb + (size_t)(b * NN + m0 + r) * NC;
  const f16* kbase = kb + (size_t)b * NN * NC;
  const f16* vbase = vt + (size_t)b * NC * NN;
  f16* P = &Pbuf[wslot][0];

  // cache the wave's Q A-fragments (16 k-steps -> 128 VGPRs)
  v16h qf[16];
#pragma unroll
  for (int kk = 0; kk < 16; ++kk) qf[kk] = load_afrag(qrow + kk * 32, h);

  v8f acc[16];
#pragma unroll
  for (int ct = 0; ct < 16; ++ct)
    acc[ct] = (v8f){0.f, 0.f, 0.f, 0.f, 0.f, 0.f, 0.f, 0.f};
  float mrow[8], lrow[8];
#pragma unroll
  for (int j = 0; j < 8; ++j) { mrow[j] = -3.0e38f; lrow[j] = 0.f; }

  for (int t = 0; t < 64; ++t) {
    const int key0 = t * BC;
    __syncthreads();  // previous tile fully consumed by all waves
#if USE_ASYNC_LDS
    // ---- async DMA stage (cache->LDS, tracked by ASYNCcnt, no VGPR round
    //      trip): 4096 16B chunks each for K and V; 16+16 chunks per thread.
    //      Builtin signature (from hipcc): (v4i AS1*, v4i AS3*, imm, imm).
#pragma unroll
    for (int c = 0; c < 16; ++c) {
      int chunk = tid + c * 256;  // 0..4095
      int krow = chunk >> 6, koff = chunk & 63;
      __builtin_amdgcn_global_load_async_to_lds_b128(
          (AS1 v4i*)(kbase + (size_t)(key0 + krow) * NC + koff * 8),
          (AS3 v4i*)(Klds + krow * KPAD + koff * 8), 0, 0);
      int vrow = chunk >> 3, voff = chunk & 7;
      __builtin_amdgcn_global_load_async_to_lds_b128(
          (AS1 v4i*)(vbase + (size_t)vrow * NN + key0 + voff * 8),
          (AS3 v4i*)(Vlds + vrow * VPAD + voff * 8), 0, 0);
    }
    __builtin_amdgcn_s_wait_asynccnt(0);  // drain own DMAs before signaling
#else
    // ---- fallback: 2048 32B chunks via VGPR (global_load_b128+ds_store)
#pragma unroll
    for (int c = 0; c < 8; ++c) {
      int chunk = tid + c * 256;  // 0..2047
      int krow = chunk >> 5, koff = chunk & 31;
      v16h kv = *(const v16h*)(kbase + (size_t)(key0 + krow) * NC + koff * 16);
      *(v16h*)(Klds + krow * KPAD + koff * 16) = kv;
      int vrow = chunk >> 2, voff = chunk & 3;
      v16h vv = *(const v16h*)(vbase + (size_t)vrow * NN + key0 + voff * 16);
      *(v16h*)(Vlds + vrow * VPAD + voff * 16) = vv;
    }
#endif
    // prefetch next tile's stream while this one is consumed
    __builtin_prefetch(kbase + (size_t)(key0 + BC + (tid >> 5)) * NC, 0, 3);
    __builtin_prefetch(vbase + (size_t)tid * NN + key0 + BC, 0, 3);
    __syncthreads();

    // ---- S = Q * K^T for 16 x 64 keys (scale folded into Q) ----
    v8f S[4];
#pragma unroll
    for (int c4 = 0; c4 < 4; ++c4)
      S[c4] = (v8f){0.f, 0.f, 0.f, 0.f, 0.f, 0.f, 0.f, 0.f};
#pragma unroll
    for (int kk = 0; kk < 16; ++kk) {
#pragma unroll
      for (int c4 = 0; c4 < 4; ++c4) {
        v16h bf = load_bfrag(Klds + (c4 * 16 + r) * KPAD + kk * 32, h);
        S[c4] = wmma_f16(qf[kk], bf, S[c4]);
      }
    }

    // ---- online softmax update ----
    float alpha[8];
#pragma unroll
    for (int j = 0; j < 8; ++j) {
      float mt = fmaxf(fmaxf(S[0][j], S[1][j]), fmaxf(S[2][j], S[3][j]));
      mt = red16_max(mt);
      float mn = fmaxf(mrow[j], mt);
      alpha[j] = __expf(mrow[j] - mn);
      mrow[j] = mn;
      float s = 0.f;
#pragma unroll
      for (int c4 = 0; c4 < 4; ++c4) {
        float p = __expf(S[c4][j] - mn);
        s += p;
        P[(j + 8 * h) * PPAD + c4 * 16 + r] = (f16)p;
      }
      s = red16_sum(s);
      lrow[j] = lrow[j] * alpha[j] + s;
    }
#pragma unroll
    for (int ct = 0; ct < 16; ++ct)
#pragma unroll
      for (int j = 0; j < 8; ++j) acc[ct][j] *= alpha[j];

    // ---- P A-fragments from LDS (same-wave DS ordering) ----
    const f16* pr = P + r * PPAD;
    v16h a0 = load_afrag(pr, h);       // keys 0..31
    v16h a1 = load_afrag(pr + 32, h);  // keys 32..63

    // ---- O += P * V (V tile in LDS, contiguous along keys) ----
#pragma unroll
    for (int ct = 0; ct < 16; ++ct) {
      const f16* vp = Vlds + (ch0 + ct * 16 + r) * VPAD;
      v16h b0 = load_bfrag(vp, h);
      v16h b1 = load_bfrag(vp + 32, h);
      acc[ct] = wmma_f16(a0, b0, acc[ct]);
      acc[ct] = wmma_f16(a1, b1, acc[ct]);
    }
  }

  // ---- normalize and store attn output f16 [b][n][c] ----
#pragma unroll
  for (int j = 0; j < 8; ++j) lrow[j] = 1.f / lrow[j];
#pragma unroll
  for (int ct = 0; ct < 16; ++ct)
#pragma unroll
    for (int j = 0; j < 8; ++j)
      ob[(size_t)(b * NN + m0 + j + 8 * h) * NC + ch0 + ct * 16 + r] =
          (f16)(acc[ct][j] * lrow[j]);
}

// ---------------------------------------------------------------------------
// K5: out = x + attn @ wp + bp  (f32 output)
// ---------------------------------------------------------------------------
__global__ __launch_bounds__(256) void proj_kernel(
    const f16* __restrict__ attn, const f16* __restrict__ wTp,
    const float* __restrict__ bp, const float* __restrict__ x,
    float* __restrict__ out) {
  const int tid = threadIdx.x, lane = tid & 31, r = lane & 15, h = lane >> 4;
  unsigned wid = blockIdx.x * 8u + (tid >> 5);
  const int ct = wid & 31;
  const int mt = (wid >> 5) & 255;
  const int b = wid >> 13;
  const f16* arow = attn + (size_t)(b * NN + mt * 16 + r) * NC;
  const f16* brow = wTp + (size_t)(ct * 16 + r) * NC;
  v8f acc = {0.f, 0.f, 0.f, 0.f, 0.f, 0.f, 0.f, 0.f};
#pragma unroll
  for (int kk = 0; kk < 16; ++kk) {
    v16h a = load_afrag(arow + kk * 32, h);
    v16h bf = load_bfrag(brow + kk * 32, h);
    acc = wmma_f16(a, bf, acc);
  }
  const float bn = bp[ct * 16 + r];
#pragma unroll
  for (int j = 0; j < 8; ++j) {
    size_t idx = (size_t)(b * NN + mt * 16 + j + 8 * h) * NC + ct * 16 + r;
    out[idx] = acc[j] + bn + x[idx];
  }
}

// ---------------------------------------------------------------------------
extern "C" void kernel_launch(void* const* d_in, const int* in_sizes, int n_in,
                              void* d_out, int out_size, void* d_ws,
                              size_t ws_size, hipStream_t stream) {
  const float* x = (const float*)d_in[0];
  const float* gamma = (const float*)d_in[1];
  const float* beta = (const float*)d_in[2];
  const float* wq = (const float*)d_in[3];
  const float* bq = (const float*)d_in[4];
  const float* wk = (const float*)d_in[5];
  const float* bk = (const float*)d_in[6];
  const float* wv = (const float*)d_in[7];
  const float* bv = (const float*)d_in[8];
  const float* wp = (const float*)d_in[9];
  const float* bp = (const float*)d_in[10];
  float* out = (float*)d_out;

  // workspace layout (~42 MB, L2-resident on MI455X's 192 MB L2):
  char* ws = (char*)d_ws;
  float* stats = (float*)ws;                // 4 KB reserved
  f16* wT = (f16*)(ws + 4096);              // 4 x 512 x 512 f16 = 2 MB
  f16* wTq = wT;
  f16* wTk = wT + 262144;
  f16* wTv = wT + 524288;
  f16* wTp = wT + 786432;
  f16* hn = (f16*)(ws + 4096 + (1u << 21));
  const size_t BUF = (size_t)NB * NN * NC;  // 4,194,304 elements (8 MB f16)
  f16* qb = hn + BUF;
  f16* kb = qb + BUF;
  f16* vtb = kb + BUF;
  f16* ab = vtb + BUF;

  prep_weights_kernel<<<4096, 256, 0, stream>>>(wq, wk, wv, wp, wT);
  gn_stats_kernel<<<NB * NG, 256, 0, stream>>>(x, stats);
  gn_apply_kernel<<<4096, 256, 0, stream>>>(x, gamma, beta, stats, hn);
  // waves: B * 3 matrices * (N/16) * (C/16) = 49152 -> 6144 blocks of 8 waves
  qkv_gemm_kernel<<<6144, 256, 0, stream>>>(hn, wTq, wTk, wTv, bq, bk, bv, qb,
                                            kb, vtb);
  // waves: B * (N/16) * 2 channel-halves = 1024 -> 128 blocks
  attn_kernel<<<128, 256, 0, stream>>>(qb, kb, vtb, ab);
  // waves: B * (N/16) * (C/16) = 16384 -> 2048 blocks
  proj_kernel<<<2048, 256, 0, stream>>>(ab, wTp, bp, x, out);
}
